// MultiHeadAttention_22290880266740
// MI455X (gfx1250) — matrix operand order repaired and checked by reading, not compiled, not run on hardware
//
#include <hip/hip_runtime.h>
#include <hip/hip_bf16.h>

typedef __attribute__((ext_vector_type(16))) __bf16 v16bf;
typedef __attribute__((ext_vector_type(8)))  float  v8f;
typedef unsigned short u16;
typedef unsigned int   u32;

union Frag16 { v16bf bf; u16 u[16]; };

__device__ __forceinline__ u16 f2bf(float f) {
    union { float f; u32 u; } x; x.f = f;
    u32 r = x.u + 0x7FFFu + ((x.u >> 16) & 1u);   // round-to-nearest-even
    return (u16)(r >> 16);
}

// K index for 16-bit A fragment element e (contiguous order: k = 16*half + e).
// hi8 = 8 for lanes 16..31.
__device__ __forceinline__ int mapA_k(int e, int hi8) {
    return e + (hi8 << 1);
}

// ---------------------------------------------------------------------------
// Double-buffered bf16 WMMA GEMM: C[M x 1024] = A[M x 1024] @ W[1024x1024] + b
// A fp32 (converted on the fly) or bf16; C bf16 or fp32.
// 256 threads (8 waves), tile 128x128, K-step 32, 2x LDS buffers (40 KB).
// LDS row stride 40 elems = 80B: 16B-aligned (b128 ds ops) and 20*row mod 64
// is a bijection over 16 rows -> conflict-free fragment reads.
// (Register staging kept here because fp32->bf16 conversion must pass VALU.)
// ---------------------------------------------------------------------------
template<bool A_IS_BF16, bool OUT_IS_BF16>
__global__ __launch_bounds__(256)
void gemm1024(const void* __restrict__ Aptr, const float* __restrict__ W,
              const float* __restrict__ bias, void* __restrict__ Cptr) {
    constexpr int KD = 1024;
    __shared__ u16 As[2][128][40];
    __shared__ u16 Bs[2][128][40];

    const int tid  = threadIdx.x;
    const int lane = tid & 31;
    const int wid  = tid >> 5;
    const int wm   = wid >> 2;            // 0..1 -> 64 rows each
    const int wn   = wid & 3;             // 0..3 -> 32 cols each
    const int row0 = blockIdx.y * 128;
    const int col0 = blockIdx.x * 128;
    const int l16  = lane & 15;
    const int hi8  = (lane >> 4) << 3;
    const int hi16 = (lane >> 4) << 4;

    v8f acc[4][2];
#pragma unroll
    for (int mi = 0; mi < 4; ++mi)
#pragma unroll
        for (int ni = 0; ni < 2; ++ni)
#pragma unroll
            for (int r = 0; r < 8; ++r) acc[mi][ni][r] = 0.0f;

    float4 aRegF[4];
    uint4  aRegB[2];
    float4 bReg[4];

    auto loadGlobal = [&](int kk) {
        if constexpr (!A_IS_BF16) {
            const float* Af = (const float*)Aptr;
#pragma unroll
            for (int it = 0; it < 4; ++it) {
                int idx = it * 256 + tid;
                int row = idx >> 3, c4 = idx & 7;
                aRegF[it] = *((const float4*)(Af + (size_t)(row0 + row) * KD + kk + c4 * 4));
            }
        } else {
            const u16* Ab = (const u16*)Aptr;
#pragma unroll
            for (int it = 0; it < 2; ++it) {
                int idx = it * 256 + tid;
                int row = idx >> 2, c8 = idx & 3;
                aRegB[it] = *((const uint4*)(Ab + (size_t)(row0 + row) * KD + kk + c8 * 8));
            }
        }
#pragma unroll
        for (int it = 0; it < 4; ++it) {
            int idx = it * 256 + tid;
            int k = idx >> 5, n4 = idx & 31;
            bReg[it] = *((const float4*)(W + (size_t)(kk + k) * KD + col0 + n4 * 4));
        }
    };

    auto storeLds = [&](int buf) {
        if constexpr (!A_IS_BF16) {
#pragma unroll
            for (int it = 0; it < 4; ++it) {
                int idx = it * 256 + tid;
                int row = idx >> 3, c4 = idx & 7;
                u32* p = (u32*)&As[buf][row][c4 * 4];
                p[0] = (u32)f2bf(aRegF[it].x) | ((u32)f2bf(aRegF[it].y) << 16);
                p[1] = (u32)f2bf(aRegF[it].z) | ((u32)f2bf(aRegF[it].w) << 16);
            }
        } else {
#pragma unroll
            for (int it = 0; it < 2; ++it) {
                int idx = it * 256 + tid;
                int row = idx >> 2, c8 = idx & 3;
                u32* p = (u32*)&As[buf][row][c8 * 8];   // 16B aligned -> b128
                p[0] = aRegB[it].x; p[1] = aRegB[it].y;
                p[2] = aRegB[it].z; p[3] = aRegB[it].w;
            }
        }
#pragma unroll
        for (int it = 0; it < 4; ++it) {
            int idx = it * 256 + tid;
            int k = idx >> 5, n4 = idx & 31;
            Bs[buf][n4 * 4 + 0][k] = f2bf(bReg[it].x);
            Bs[buf][n4 * 4 + 1][k] = f2bf(bReg[it].y);
            Bs[buf][n4 * 4 + 2][k] = f2bf(bReg[it].z);
            Bs[buf][n4 * 4 + 3][k] = f2bf(bReg[it].w);
        }
    };

    loadGlobal(0);
    for (int i = 0; i < KD / 32; ++i) {
        const int buf = i & 1;
        storeLds(buf);
        __syncthreads();
        if (i + 1 < KD / 32) {
            loadGlobal((i + 1) * 32);   // latency hides under the WMMAs below
            if (i + 2 < KD / 32)        // L2 prefetch of the tile after next
                __builtin_prefetch((const void*)(W + (size_t)((i + 2) * 32 + (tid >> 3)) * KD
                                                   + col0 + (tid & 7) * 16), 0, 3);
        }

        Frag16 aF[4], bF[2];
#pragma unroll
        for (int mi = 0; mi < 4; ++mi) {
            int row = wm * 64 + mi * 16 + l16;
#pragma unroll
            for (int e = 0; e < 16; ++e) aF[mi].u[e] = As[buf][row][mapA_k(e, hi8)];
        }
#pragma unroll
        for (int ni = 0; ni < 2; ++ni) {
            int n = wn * 32 + ni * 16 + l16;
#pragma unroll
            for (int e = 0; e < 16; ++e) bF[ni].u[e] = Bs[buf][n][e + hi16];
        }
#pragma unroll
        for (int mi = 0; mi < 4; ++mi)
#pragma unroll
            for (int ni = 0; ni < 2; ++ni)
                acc[mi][ni] = __builtin_amdgcn_wmma_f32_16x16x32_bf16(
                    false, aF[mi].bf, false, bF[ni].bf, (short)0, acc[mi][ni], false, false);
        // single barrier per step: next iteration writes the *other* buffer
    }

#pragma unroll
    for (int mi = 0; mi < 4; ++mi) {
#pragma unroll
        for (int ni = 0; ni < 2; ++ni) {
            int n = col0 + wn * 32 + ni * 16 + l16;
            float bvs = bias[n];
#pragma unroll
            for (int r = 0; r < 8; ++r) {
                int m = row0 + wm * 64 + mi * 16 + r + hi8;
                float v = acc[mi][ni][r] + bvs;
                if constexpr (OUT_IS_BF16)
                    ((u16*)Cptr)[(size_t)m * 1024 + n] = f2bf(v);
                else
                    ((float*)Cptr)[(size_t)m * 1024 + n] = v;
            }
        }
    }
}

// ---------------------------------------------------------------------------
// Per-token HxH attention. One wave per token; 4 waves per block; every LDS
// region is per-wave, so no workgroup barriers: same-wave LDS ops are in-order
// and s_wait_dscnt / s_wait_asynccnt provide the completion guarantees.
// Q/K/V are already bf16, so staging uses GLOBAL_LOAD_ASYNC_TO_LDS_B128
// (ASYNCcnt path, no VGPR round-trip); ctx goes back out with
// GLOBAL_STORE_ASYNC_FROM_LDS_B128.
// ---------------------------------------------------------------------------
__global__ __launch_bounds__(128)
void attn_kernel(const u16* __restrict__ Q, const u16* __restrict__ Kp,
                 const u16* __restrict__ V, u16* __restrict__ CTX) {
    __shared__ u16  qs[4][16][72];   // 144B stride: 16B aligned, conflict-free
    __shared__ u16  ks[4][16][72];
    __shared__ u16  vs[4][16][72];
    __shared__ float smat[4][16][17]; // f32 score tile for softmax relayout
    __shared__ u16  wsm[4][16][18];
    __shared__ u16  os[4][16][72];

    const int tid  = threadIdx.x;
    const int lane = tid & 31;
    const int w    = tid >> 5;
    const int tok  = blockIdx.x * 4 + w;
    const int b    = tok >> 12;
    const int n    = tok & 4095;
    const int rq   = n >> 4;
    const int cb   = (n & 15) << 6;
    const size_t base = ((size_t)b * 4096 + rq) * 1024 + cb;   // + h*256*1024

    // ---- async global->LDS staging: 16 rows x 64 bf16 per matrix ----
    {
        int r0 = lane >> 3;
        int c8 = lane & 7;
#pragma unroll
        for (int it = 0; it < 4; ++it) {
            int h = it * 4 + r0;
            u32 goff = (u32)((base + (size_t)h * 256 * 1024 + c8 * 8) * 2); // bytes
            u32 ldq = (u32)(size_t)&qs[w][h][c8 * 8];
            u32 ldk = (u32)(size_t)&ks[w][h][c8 * 8];
            u32 ldv = (u32)(size_t)&vs[w][h][c8 * 8];
            asm volatile("global_load_async_to_lds_b128 %0, %1, %2"
                         :: "v"(ldq), "v"(goff), "s"(Q)  : "memory");
            asm volatile("global_load_async_to_lds_b128 %0, %1, %2"
                         :: "v"(ldk), "v"(goff), "s"(Kp) : "memory");
            asm volatile("global_load_async_to_lds_b128 %0, %1, %2"
                         :: "v"(ldv), "v"(goff), "s"(V)  : "memory");
        }
    }
    asm volatile("s_wait_asynccnt 0x0" ::: "memory");

    const int l16  = lane & 15;
    const int hi8  = (lane >> 4) << 3;
    const int hi16 = (lane >> 4) << 4;

    // ---- scores = q @ k^T (K=64): two chained WMMAs ----
    v8f sc;
#pragma unroll
    for (int r = 0; r < 8; ++r) sc[r] = 0.0f;
#pragma unroll
    for (int kc = 0; kc < 2; ++kc) {
        Frag16 a, bfr;
#pragma unroll
        for (int e = 0; e < 16; ++e) a.u[e]   = qs[w][l16][kc * 32 + mapA_k(e, hi8)];
#pragma unroll
        for (int e = 0; e < 16; ++e) bfr.u[e] = ks[w][l16][kc * 32 + e + hi16];
        sc = __builtin_amdgcn_wmma_f32_16x16x32_bf16(
            false, a.bf, false, bfr.bf, (short)0, sc, false, false);
    }

    // ---- softmax over g: LDS relayout, serial reduce, 1 shfl_xor ----
#pragma unroll
    for (int r = 0; r < 8; ++r) smat[w][r + hi8][l16] = sc[r] * 0.125f; // 1/sqrt(64)
    asm volatile("s_wait_dscnt 0x0" ::: "memory");   // same-wave RAW via LDS
    {
        const int row  = l16;
        const int half = (lane >> 4) << 3;           // 0 or 8: this lane's 8 cols
        float v0[8];
#pragma unroll
        for (int j = 0; j < 8; ++j) v0[j] = smat[w][row][half + j];
        float m = v0[0];
#pragma unroll
        for (int j = 1; j < 8; ++j) m = fmaxf(m, v0[j]);
        m = fmaxf(m, __shfl_xor(m, 16, 32));         // combine the two halves
        float p[8], s = 0.0f;
#pragma unroll
        for (int j = 0; j < 8; ++j) { p[j] = __expf(v0[j] - m); s += p[j]; }
        s += __shfl_xor(s, 16, 32);
        float invs = 1.0f / s;
#pragma unroll
        for (int j = 0; j < 8; ++j) wsm[w][row][half + j] = f2bf(p[j] * invs);
    }
    asm volatile("s_wait_dscnt 0x0" ::: "memory");

    // ---- o = w @ v : A = weights (K padded 16->32), B = v chunks ----
    Frag16 aw;
#pragma unroll
    for (int e = 0; e < 16; ++e) {
        int k = mapA_k(e, hi8);
        aw.u[e] = (k < 16) ? wsm[w][l16][k] : (u16)0;
    }
#pragma unroll
    for (int c4 = 0; c4 < 4; ++c4) {
        Frag16 bv;
#pragma unroll
        for (int e = 0; e < 16; ++e) {
            int k = e + hi16;
            bv.u[e] = (k < 16) ? vs[w][k][c4 * 16 + l16] : (u16)0;
        }
        v8f o;
#pragma unroll
        for (int r = 0; r < 8; ++r) o[r] = 0.0f;
        o = __builtin_amdgcn_wmma_f32_16x16x32_bf16(
            false, aw.bf, false, bv.bf, (short)0, o, false, false);
#pragma unroll
        for (int r = 0; r < 8; ++r) os[w][r + hi8][c4 * 16 + l16] = f2bf(o[r]);
    }
    asm volatile("s_wait_dscnt 0x0" ::: "memory");   // os visible to async engine

    // ---- async LDS->global store back to the scattered ctx layout ----
    {
        int r0 = lane >> 3;
        int c8 = lane & 7;
#pragma unroll
        for (int it = 0; it < 4; ++it) {
            int h = it * 4 + r0;
            u32 goff = (u32)((base + (size_t)h * 256 * 1024 + c8 * 8) * 2); // bytes
            u32 ldo = (u32)(size_t)&os[w][h][c8 * 8];
            asm volatile("global_store_async_from_lds_b128 %0, %1, %2"
                         :: "v"(goff), "v"(ldo), "s"(CTX) : "memory");
        }
    }
    asm volatile("s_wait_asynccnt 0x0" ::: "memory");
}

// ---------------------------------------------------------------------------
extern "C" void kernel_launch(void* const* d_in, const int* in_sizes, int n_in,
                              void* d_out, int out_size, void* d_ws, size_t ws_size,
                              hipStream_t stream) {
    const float* query = (const float*)d_in[0];
    const float* key_  = (const float*)d_in[1];
    const float* value = (const float*)d_in[2];
    const float* Wq = (const float*)d_in[3];
    const float* bq = (const float*)d_in[4];
    const float* Wk = (const float*)d_in[5];
    const float* bk = (const float*)d_in[6];
    const float* Wv = (const float*)d_in[7];
    const float* bv = (const float*)d_in[8];
    const float* Wo = (const float*)d_in[9];
    const float* bo = (const float*)d_in[10];

    const size_t MAT = (size_t)16384 * 1024;     // elements per projection matrix
    u16* Qb  = (u16*)d_ws;                       // bf16 intermediates in workspace
    u16* Kb  = Qb + MAT;
    u16* Vb  = Kb + MAT;
    u16* CTX = Vb + MAT;

    dim3 gG(8, 128, 1), bG(256, 1, 1);
    gemm1024<false, true><<<gG, bG, 0, stream>>>((const void*)query, Wq, bq, (void*)Qb);
    gemm1024<false, true><<<gG, bG, 0, stream>>>((const void*)key_,  Wk, bk, (void*)Kb);
    gemm1024<false, true><<<gG, bG, 0, stream>>>((const void*)value, Wv, bv, (void*)Vb);

    attn_kernel<<<dim3(4096, 1, 1), dim3(128, 1, 1), 0, stream>>>(Qb, Kb, Vb, CTX);

    gemm1024<true, false><<<gG, bG, 0, stream>>>((const void*)CTX, Wo, bo, d_out);
}
